// GraniteMoeHybridParallelExpertsFP8_79156247265584
// MI455X (gfx1250) — matrix operand order repaired and checked by reading, not compiled, not run on hardware
//
#include <hip/hip_runtime.h>
#include <hip/hip_bf16.h>

// Problem constants (match reference)
#define NUM_EXPERTS 8
#define IN_K        2048
#define OUT_N       8192
#define TOK_M       1024

// Tiling: 128(M) x 256(N) block tile, 512 threads = 16 wave32 in a 2x8 grid,
// each wave owns a 64x32 sub-tile = 4x2 WMMA accumulators.
#define BM 128
#define BN 256
#define KT 32            // K chunk = one bf16 WMMA K-step
#define AST 40           // LDS row stride (bf16 elems): 80B rows -> 16B aligned + bank-conflict pad
#define BST 40
#define NCHUNK (IN_K / KT)   // 64
#define NTHREADS 512

typedef __attribute__((ext_vector_type(16))) __bf16 v16bf;
typedef __attribute__((ext_vector_type(8)))  float  v8f;

union Frag16 { v16bf v; uint4 q[2]; };

// f32x2 -> packed bf16x2 (round-half-up) in exactly 3 VALU ops:
// two v_add_nc_u32 + one v_perm_b32 extracting the two high halves.
static __device__ __forceinline__ unsigned int pack_bf16x2(float lo, float hi) {
    unsigned int rlo = __float_as_uint(lo) + 0x8000u;
    unsigned int rhi = __float_as_uint(hi) + 0x8000u;
    // v_perm_b32: bytes 0-3 select from S1 (rlo), 4-7 from S0 (rhi).
    // 0x07060302 -> [rhi.b3, rhi.b2, rlo.b3, rlo.b2] = {bf16(hi), bf16(lo)}
    return __builtin_amdgcn_perm(rhi, rlo, 0x07060302u);
}

__global__ __launch_bounds__(NTHREADS)
void GraniteMoeFP8GroupedGemm_wmma_bf16(const float* __restrict__ x,
                                        const float* __restrict__ w,
                                        const float* __restrict__ wscale,
                                        float* __restrict__ out)
{
    __shared__ unsigned short Ash[2][BM * AST];
    __shared__ unsigned short Bsh[2][BN * BST];

    const int tid  = threadIdx.x;
    const int lane = tid & 31;
    const int wid  = tid >> 5;     // 0..15
    const int wm   = wid & 1;      // 0..1  (64 rows each)
    const int wn   = wid >> 1;     // 0..7  (32 cols each)

    const int bid = blockIdx.x;
    const int e   = bid >> 8;          // 256 blocks per expert (8 m-tiles * 32 n-tiles)
    const int mt  = (bid >> 5) & 7;    // token tile within expert
    const int nt  = bid & 31;          // output tile within expert

    const size_t abase = ((size_t)e * TOK_M + (size_t)mt * BM) * IN_K;
    const size_t bbase = ((size_t)e * OUT_N + (size_t)nt * BN) * IN_K;

    // Fill mapping (f32 -> float4 granularity, 8 float4 per 32-wide row):
    //   A tile: 128 rows x 8 f4 = 1024 f4 -> 2 per thread
    //   B tile: 256 rows x 8 f4 = 2048 f4 -> 4 per thread
    const int frow = tid >> 3;     // 0..63, A: +i*64 ; B: +i*64
    const int fc4  = tid & 7;      // float4 column

    float4 ar[2], br[4];

    auto fetch = [&](int c) {
        const size_t kc = (size_t)c * KT + (size_t)fc4 * 4;
#pragma unroll
        for (int i = 0; i < 2; ++i)
            ar[i] = *(const float4*)(x + abase + (size_t)(frow + i * 64) * IN_K + kc);
#pragma unroll
        for (int i = 0; i < 4; ++i)
            br[i] = *(const float4*)(w + bbase + (size_t)(frow + i * 64) * IN_K + kc);
    };
    auto stage = [&](int buf) {
#pragma unroll
        for (int i = 0; i < 2; ++i) {
            uint2 pa = make_uint2(pack_bf16x2(ar[i].x, ar[i].y), pack_bf16x2(ar[i].z, ar[i].w));
            *(uint2*)&Ash[buf][(frow + i * 64) * AST + fc4 * 4] = pa;
        }
#pragma unroll
        for (int i = 0; i < 4; ++i) {
            uint2 pb = make_uint2(pack_bf16x2(br[i].x, br[i].y), pack_bf16x2(br[i].z, br[i].w));
            *(uint2*)&Bsh[buf][(frow + i * 64) * BST + fc4 * 4] = pb;
        }
    };

    v8f acc[4][2];
    const v8f vzero = {0.f, 0.f, 0.f, 0.f, 0.f, 0.f, 0.f, 0.f};
#pragma unroll
    for (int i = 0; i < 4; ++i)
#pragma unroll
        for (int j = 0; j < 2; ++j) acc[i][j] = vzero;

    // A layout (matches B): lanes 0-15 -> K[0..16), lanes 16-31 -> K[16..32)
    const int arow = wm * 64 + (lane & 15);
    const int akb  = (lane < 16) ? 0 : 16;
    // B 32x16: N = lane&15, lanes 0-15 hold K[0..16), lanes 16-31 hold K[16..32)
    const int bcol = wn * 32 + (lane & 15);
    const int bkb  = (lane < 16) ? 0 : 16;

    fetch(0);
    stage(0);
    __syncthreads();

    for (int c = 0; c < NCHUNK; ++c) {
        const int buf = c & 1;

        if (c + 1 < NCHUNK) {
            fetch(c + 1);
            if (c + 2 < NCHUNK) {   // gfx1250 global_prefetch_b8 two chunks ahead
                const size_t kc2 = (size_t)(c + 2) * KT + (size_t)fc4 * 4;
                __builtin_prefetch(x + abase + (size_t)frow * IN_K + kc2, 0, 2);
                __builtin_prefetch(w + bbase + (size_t)frow * IN_K + kc2, 0, 2);
            }
        }

        Frag16 af[4], bf[2];
#pragma unroll
        for (int fm = 0; fm < 4; ++fm) {
            const unsigned short* p = &Ash[buf][(arow + fm * 16) * AST + akb];
            af[fm].q[0] = *(const uint4*)(p);
            af[fm].q[1] = *(const uint4*)(p + 8);
        }
#pragma unroll
        for (int fn = 0; fn < 2; ++fn) {
            const unsigned short* p = &Bsh[buf][(bcol + fn * 16) * BST + bkb];
            bf[fn].q[0] = *(const uint4*)(p);
            bf[fn].q[1] = *(const uint4*)(p + 8);
        }

#pragma unroll
        for (int fm = 0; fm < 4; ++fm)
#pragma unroll
            for (int fn = 0; fn < 2; ++fn)
                acc[fm][fn] = __builtin_amdgcn_wmma_f32_16x16x32_bf16(
                    false, af[fm].v, false, bf[fn].v,
                    (short)0, acc[fm][fn], false, false);

        if (c + 1 < NCHUNK) stage((c + 1) & 1);
        __syncthreads();
    }

    // Epilogue: apply per-output-row dequant scale on the f32 accumulator.
    // C/D layout: VGPR r -> M = r + (lane<16 ? 0 : 8), N = lane&15.
    const int colbase = nt * BN + wn * 32 + (lane & 15);
    const int rowm    = mt * BM + wm * 64 + ((lane < 16) ? 0 : 8);
#pragma unroll
    for (int fn = 0; fn < 2; ++fn) {
        const int o   = colbase + fn * 16;
        const float s = wscale[(size_t)e * OUT_N + o];
#pragma unroll
        for (int fm = 0; fm < 4; ++fm) {
            const size_t base = ((size_t)e * TOK_M + (size_t)(rowm + fm * 16)) * OUT_N + o;
#pragma unroll
            for (int r = 0; r < 8; ++r)
                out[base + (size_t)r * OUT_N] = acc[fm][fn][r] * s;
        }
    }
}

extern "C" void kernel_launch(void* const* d_in, const int* in_sizes, int n_in,
                              void* d_out, int out_size, void* d_ws, size_t ws_size,
                              hipStream_t stream) {
    const float* x      = (const float*)d_in[0];
    // d_in[1] = expert_size (int32); equal static split -> unused for addressing
    const float* w      = (const float*)d_in[2];
    const float* wscale = (const float*)d_in[3];
    float* out = (float*)d_out;
    (void)in_sizes; (void)n_in; (void)out_size; (void)d_ws; (void)ws_size;

    const int nblocks = NUM_EXPERTS * (TOK_M / BM) * (OUT_N / BN);  // 2048
    GraniteMoeFP8GroupedGemm_wmma_bf16<<<nblocks, NTHREADS, 0, stream>>>(x, w, wscale, out);
}
